// SparseDeformableChannelMambaBlock_39633958208118
// MI455X (gfx1250) — matrix operand order repaired and checked by reading, not compiled, not run on hardware
//
#include <hip/hip_runtime.h>

// ---------------------------------------------------------------------------
// SparseDeformableChannelMambaBlock for MI455X (gfx1250, wave32, WMMA).
// GEMMs: v_wmma_f32_16x16x32_bf16 with hi/lo bf16 split (3 products) for
// ~fp32 accuracy at bf16-pipe rate; 2x2 tile register blocking per wave
// (12 WMMA per 16 b128 loads per 32-K chunk).
// Bulk global->LDS staging uses gfx1250 async loads + s_wait_asynccnt.
// ---------------------------------------------------------------------------

#define BB   32
#define LL   4096
#define DD   256
#define EE   336
#define SS   16
#define NHH  6
#define HDD  56
#define KCH  1228
#define NTOK (BB * LL)          // 131072 tokens
#define EP   352                // E padded to multiple of 32 (WMMA K-chunk)
#define Q3   1008               // 3*E
#define MR2  (BB * KCH)         // 39296 rows for output GEMM (= 128 * 307)

#define USE_ASYNC_LDS 1

typedef __attribute__((ext_vector_type(16))) __bf16 v16bf;
typedef __attribute__((ext_vector_type(8)))  float  v8f;

union frag_u { v16bf v; unsigned short s[16]; };

// ---- bf16 storage helpers (bit-level, RNE rounding) ------------------------
__device__ __forceinline__ unsigned short f2bf(float x) {
    union { float f; unsigned u; } c; c.f = x;
    unsigned r = c.u + 0x7FFFu + ((c.u >> 16) & 1u);
    return (unsigned short)(r >> 16);
}
__device__ __forceinline__ float bf2f(unsigned short h) {
    union { unsigned u; float f; } c; c.u = ((unsigned)h) << 16;
    return c.f;
}

// ---- gfx1250 async global->LDS helpers (inline asm; ISA ch.10 / 15.18) -----
#if USE_ASYNC_LDS
__device__ __forceinline__ void async_load_b128(unsigned lds_byte_off, const void* gptr) {
    asm volatile("global_load_async_to_lds_b128 %0, %1, off"
                 :: "v"(lds_byte_off), "v"(gptr)
                 : "memory");
}
__device__ __forceinline__ void wait_async0() {
    asm volatile("s_wait_asynccnt 0" ::: "memory");
}
// low 32 bits of a generic pointer to __shared__ == LDS byte address
__device__ __forceinline__ unsigned lds_off(const void* p) {
    return (unsigned)(unsigned long long)p;
}
#endif

// ---- WMMA fragment loads (wave32 layouts, ISA 7.12.2) ----------------------
// A 16x32 bf16: lane half 'hf', element j -> K = 8*hf + j (j<8), 16+8*hf+(j-8)
__device__ __forceinline__ v16bf frag_ld_a(const unsigned short* __restrict__ rowp, int hf) {
    frag_u f;
    const unsigned short* p0 = rowp + 8 * hf;
    const unsigned short* p1 = rowp + 16 + 8 * hf;
#pragma unroll
    for (int j = 0; j < 8; ++j) { f.s[j] = p0[j]; f.s[8 + j] = p1[j]; }
    return f.v;
}
// B 32x16 bf16: lane = column, element j -> K = 8*hf + j (j<8), 16+8*hf+(j-8)
__device__ __forceinline__ v16bf frag_ld_b(const unsigned short* __restrict__ rowp, int hf) {
    frag_u f;
    const unsigned short* p0 = rowp + 8 * hf;
    const unsigned short* p1 = rowp + 16 + 8 * hf;
#pragma unroll
    for (int j = 0; j < 8; ++j) { f.s[j] = p0[j]; f.s[8 + j] = p1[j]; }
    return f.v;
}

__device__ __forceinline__ v8f wmma3(v8f acc, v16bf ah, v16bf al, v16bf bh, v16bf bl) {
    acc = __builtin_amdgcn_wmma_f32_16x16x32_bf16(false, ah, false, bh, (short)0, acc, false, false);
    acc = __builtin_amdgcn_wmma_f32_16x16x32_bf16(false, ah, false, bl, (short)0, acc, false, false);
    acc = __builtin_amdgcn_wmma_f32_16x16x32_bf16(false, al, false, bh, (short)0, acc, false, false);
    return acc;
}

// ---- shared 2x2-blocked mainloop: wave computes 32x32 output ---------------
__device__ __forceinline__ void gemm_mainloop_2x2(
    const unsigned short* __restrict__ Ahi, const unsigned short* __restrict__ Alo, int lda,
    const unsigned short* __restrict__ Bhi, const unsigned short* __restrict__ Blo, int ldb,
    int r0, int c0, int li, int hf, int Kp, v8f (&acc)[2][2]) {
    const unsigned short* a0h = Ahi + (size_t)(r0 + li) * lda;
    const unsigned short* a0l = Alo + (size_t)(r0 + li) * lda;
    const unsigned short* a1h = Ahi + (size_t)(r0 + 16 + li) * lda;
    const unsigned short* a1l = Alo + (size_t)(r0 + 16 + li) * lda;
    const unsigned short* b0h = Bhi + (size_t)(c0 + li) * ldb;
    const unsigned short* b0l = Blo + (size_t)(c0 + li) * ldb;
    const unsigned short* b1h = Bhi + (size_t)(c0 + 16 + li) * ldb;
    const unsigned short* b1l = Blo + (size_t)(c0 + 16 + li) * ldb;
    for (int kb = 0; kb < Kp; kb += 32) {
        v16bf fa0h = frag_ld_a(a0h + kb, hf);
        v16bf fa0l = frag_ld_a(a0l + kb, hf);
        v16bf fa1h = frag_ld_a(a1h + kb, hf);
        v16bf fa1l = frag_ld_a(a1l + kb, hf);
        v16bf fb0h = frag_ld_b(b0h + kb, hf);
        v16bf fb0l = frag_ld_b(b0l + kb, hf);
        v16bf fb1h = frag_ld_b(b1h + kb, hf);
        v16bf fb1l = frag_ld_b(b1l + kb, hf);
        acc[0][0] = wmma3(acc[0][0], fa0h, fa0l, fb0h, fb0l);
        acc[0][1] = wmma3(acc[0][1], fa0h, fa0l, fb1h, fb1l);
        acc[1][0] = wmma3(acc[1][0], fa1h, fa1l, fb0h, fb0l);
        acc[1][1] = wmma3(acc[1][1], fa1h, fa1l, fb1h, fb1l);
    }
}

// ---------------------------------------------------------------------------
// K-pack: fp32 (n x k) -> hi/lo bf16 (npad x kp); zero pad rows [n,npad) and
// cols [k,kp). Padded rows/cols multiply as zeros -> exact results.
// ---------------------------------------------------------------------------
__global__ void pack_w(const float* __restrict__ W, unsigned short* __restrict__ hi,
                       unsigned short* __restrict__ lo, int n, int npad, int k, int kp) {
    long i = (long)blockIdx.x * blockDim.x + threadIdx.x;
    if (i >= (long)npad * kp) return;
    int r = (int)(i / kp), c = (int)(i % kp);
    float v = (r < n && c < k) ? W[(size_t)r * k + c] : 0.f;
    unsigned short h = f2bf(v);
    hi[i] = h;
    lo[i] = f2bf(v - bf2f(h));
}

// ---------------------------------------------------------------------------
// DyT: xn = tanh(alpha*x)*w + b -> hi/lo bf16 (K=256, no padding needed)
// ---------------------------------------------------------------------------
__global__ void dyt_split(const float* __restrict__ x, const float* __restrict__ alpha,
                          const float* __restrict__ w, const float* __restrict__ b,
                          unsigned short* __restrict__ hi, unsigned short* __restrict__ lo) {
    long i = (long)blockIdx.x * blockDim.x + threadIdx.x;
    if (i >= (long)NTOK * DD) return;
    int c = (int)(i & (DD - 1));
    float v = tanhf(alpha[0] * x[i]) * w[c] + b[c];
    unsigned short h = f2bf(v);
    hi[i] = h;
    lo[i] = f2bf(v - bf2f(h));
}

// ---------------------------------------------------------------------------
// GEMM: C(MxN) = Abf(MxKp) * Bbf(NpadxKp)^T + bias, fp32 out.
// Block = 8 waves as 4x2 -> 128 rows x 64 cols. grid = (Npad/64, M/128).
// Stores guarded by col < N (WMMA itself always runs with EXEC all ones).
// ---------------------------------------------------------------------------
__global__ __launch_bounds__(256) void gemm_bias_f32(
    const unsigned short* __restrict__ Ahi, const unsigned short* __restrict__ Alo, int lda,
    const unsigned short* __restrict__ Bhi, const unsigned short* __restrict__ Blo, int ldb,
    const float* __restrict__ bias, float* __restrict__ C, int ldc, int N, int Kp) {
    int w = threadIdx.x >> 5, lane = threadIdx.x & 31;
    int wr = w >> 1, wc = w & 1;
    int hf = lane >> 4, li = lane & 15;
    int r0 = blockIdx.y * 128 + wr * 32;
    int c0 = blockIdx.x * 64 + wc * 32;
    v8f acc[2][2] = {};
    gemm_mainloop_2x2(Ahi, Alo, lda, Bhi, Blo, ldb, r0, c0, li, hf, Kp, acc);
#pragma unroll
    for (int i = 0; i < 2; ++i)
#pragma unroll
        for (int j = 0; j < 2; ++j) {
            int col = c0 + j * 16 + li;
            if (col < N) {
                float bc = bias[col];
#pragma unroll
                for (int r = 0; r < 8; ++r) {
                    int row = r0 + i * 16 + hf * 8 + r;   // C/D: vgpr r -> row r + 8*half
                    C[(size_t)row * ldc + col] = acc[i][j][r] + bc;
                }
            }
        }
}

// ---------------------------------------------------------------------------
// Row L2-normalize xp_raw (NTOK x 336) -> hi/lo bf16 padded to EP. Wave/row.
// ---------------------------------------------------------------------------
__global__ void norm_split(const float* __restrict__ xp, unsigned short* __restrict__ hi,
                           unsigned short* __restrict__ lo) {
    int lane = threadIdx.x & 31;
    long row = (long)blockIdx.x * (blockDim.x >> 5) + (threadIdx.x >> 5);
    if (row >= NTOK) return;
    const float* p = xp + row * EE;
    float ss = 0.f;
    for (int c = lane; c < EE; c += 32) { float v = p[c]; ss = fmaf(v, v, ss); }
#pragma unroll
    for (int m = 16; m >= 1; m >>= 1) ss += __shfl_xor(ss, m);
    float scale = 1.f / fmaxf(sqrtf(ss), 1e-12f);
    for (int c = lane; c < EP; c += 32) {
        float v = (c < EE) ? p[c] * scale : 0.f;
        unsigned short h = f2bf(v);
        hi[row * EP + c] = h;
        lo[row * EP + c] = f2bf(v - bf2f(h));
    }
}

// ---------------------------------------------------------------------------
// Head selection: top-1 head per (token, h) -> gather v row; imp = |out|^2
// ---------------------------------------------------------------------------
__global__ void select_heads(const float* __restrict__ qkv, float* __restrict__ attn,
                             float* __restrict__ imp) {
    int n = blockIdx.x * blockDim.x + threadIdx.x;
    if (n >= NTOK) return;
    const float* q = qkv + (size_t)n * Q3;
    const float* k = q + EE;
    const float* v = q + 2 * EE;
    float sc[NHH][NHH];
#pragma unroll
    for (int h = 0; h < NHH; ++h)
#pragma unroll
        for (int g = 0; g < NHH; ++g) sc[h][g] = 0.f;
    for (int d = 0; d < HDD; ++d) {
        float qd[NHH], kd[NHH];
#pragma unroll
        for (int h = 0; h < NHH; ++h) { qd[h] = q[h * HDD + d]; kd[h] = k[h * HDD + d]; }
#pragma unroll
        for (int h = 0; h < NHH; ++h)
#pragma unroll
            for (int g = 0; g < NHH; ++g) sc[h][g] = fmaf(qd[h], kd[g], sc[h][g]);
    }
    float ss = 0.f;
#pragma unroll
    for (int h = 0; h < NHH; ++h) {
        int gi = 0; float best = sc[h][0];
#pragma unroll
        for (int g = 1; g < NHH; ++g)
            if (sc[h][g] > best) { best = sc[h][g]; gi = g; }   // ties -> lowest g
        const float* vs = v + gi * HDD;
        float* o = attn + (size_t)n * EE + h * HDD;
        for (int d = 0; d < HDD; ++d) { float t = vs[d]; o[d] = t; ss = fmaf(t, t, ss); }
    }
    imp[n] = ss;
}

// ---------------------------------------------------------------------------
// Per-batch sorted top-K_CH (descending, ties -> lower index): bitonic sort of
// 64-bit keys in LDS. imp staged global->LDS via gfx1250 async loads.
// ---------------------------------------------------------------------------
__global__ __launch_bounds__(1024) void topk_sort(const float* __restrict__ imp,
                                                  int* __restrict__ ch_idx) {
    __shared__ float stage[LL];
    __shared__ unsigned long long key[LL];
    int b = blockIdx.x, tid = threadIdx.x;
#if USE_ASYNC_LDS
    async_load_b128(lds_off(&stage[tid * 4]), imp + (size_t)b * LL + tid * 4);
    wait_async0();
    __syncthreads();
#else
    for (int i = tid; i < LL; i += 1024) stage[i] = imp[(size_t)b * LL + i];
    __syncthreads();
#endif
    for (int i = tid; i < LL; i += 1024) {
        union { float f; unsigned u; } c; c.f = stage[i];   // imp >= 0
        key[i] = ((unsigned long long)c.u << 32) | (unsigned)(LL - 1 - i);
    }
    __syncthreads();
    for (int k = 2; k <= LL; k <<= 1) {
        for (int j = k >> 1; j > 0; j >>= 1) {
            for (int t = tid; t < LL; t += 1024) {
                int l = t ^ j;
                if (l > t) {
                    unsigned long long a = key[t], c = key[l];
                    bool descending = ((t & k) == 0);
                    if (descending ? (a < c) : (a > c)) { key[t] = c; key[l] = a; }
                }
            }
            __syncthreads();
        }
    }
    for (int i = tid; i < KCH; i += 1024)
        ch_idx[(size_t)b * KCH + i] = (LL - 1) - (int)(key[i] & 0xFFFFFFFFu);
}

// ---------------------------------------------------------------------------
// Gather selected rows + depthwise conv (kernel 4, left context 3).
// ---------------------------------------------------------------------------
__global__ void gather_conv(const float* __restrict__ attn, const int* __restrict__ ch_idx,
                            const float* __restrict__ cw, float* __restrict__ xc) {
    int e = threadIdx.x;
    if (e >= EE) return;
    int t = blockIdx.x, b = blockIdx.y;
    float acc = 0.f;
#pragma unroll
    for (int j = 0; j < 4; ++j) {
        int tt = t + j - 3;
        if (tt >= 0) {
            int l = ch_idx[(size_t)b * KCH + tt];
            acc = fmaf(cw[e * 4 + j], attn[((size_t)b * LL + l) * EE + e], acc);
        }
    }
    xc[((size_t)b * KCH + t) * EE + e] = acc;
}

// ---------------------------------------------------------------------------
// SSM scan: thread per (b, e); h' = h A^T + sigmoid(Bp)*x; y = <h', sigC[e]>.
// A staged via async load to LDS. Emits outs as hi/lo bf16 padded to EP.
// ---------------------------------------------------------------------------
__global__ __launch_bounds__(256) void ssm_scan(const float* __restrict__ xc,
                                                const float* __restrict__ Amat,
                                                const float* __restrict__ Bp,
                                                const float* __restrict__ Cp,
                                                unsigned short* __restrict__ ohi,
                                                unsigned short* __restrict__ olo) {
    __shared__ float As[SS][SS];
    __shared__ float sB[SS];
#if USE_ASYNC_LDS
    if (threadIdx.x < (SS * SS) / 4)
        async_load_b128(lds_off(&As[0][0] + threadIdx.x * 4), Amat + threadIdx.x * 4);
    wait_async0();
#else
    if (threadIdx.x < SS * SS) As[threadIdx.x >> 4][threadIdx.x & 15] = Amat[threadIdx.x];
#endif
    if (threadIdx.x < SS) sB[threadIdx.x] = 1.f / (1.f + expf(-Bp[threadIdx.x]));
    __syncthreads();
    int id = blockIdx.x * blockDim.x + threadIdx.x;
    if (id >= BB * EP) return;
    int b = id / EP, ep = id % EP;
    if (ep >= EE) {                                // pad columns: zeros
        for (int t = 0; t < KCH; ++t) {
            size_t o = ((size_t)b * KCH + t) * EP + ep;
            ohi[o] = 0; olo[o] = 0;
        }
        return;
    }
    int e = ep;
    float sC[SS];
#pragma unroll
    for (int s = 0; s < SS; ++s) sC[s] = 1.f / (1.f + expf(-Cp[e * SS + s]));
    float h[SS];
#pragma unroll
    for (int s = 0; s < SS; ++s) h[s] = 0.f;
    for (int t = 0; t < KCH; ++t) {
        float xv = xc[((size_t)b * KCH + t) * EE + e];
        float hn[SS];
#pragma unroll
        for (int tt = 0; tt < SS; ++tt) {
            float a = sB[tt] * xv;
#pragma unroll
            for (int s = 0; s < SS; ++s) a = fmaf(As[tt][s], h[s], a);
            hn[tt] = a;
        }
        float o = 0.f;
#pragma unroll
        for (int tt = 0; tt < SS; ++tt) { h[tt] = hn[tt]; o = fmaf(hn[tt], sC[tt], o); }
        size_t off = ((size_t)b * KCH + t) * EP + e;
        unsigned short hi = f2bf(o);
        ohi[off] = hi;
        olo[off] = f2bf(o - bf2f(hi));
    }
}

// ---------------------------------------------------------------------------
// Residual copy: out = x
// ---------------------------------------------------------------------------
__global__ void copy_residual(const float* __restrict__ x, float* __restrict__ out) {
    long i = (long)blockIdx.x * blockDim.x + threadIdx.x;
    if (i < (long)NTOK * DD) out[i] = x[i];
}

// ---------------------------------------------------------------------------
// Output GEMM (MR2 x 256, K=EP) + scatter: out[b, ch_idx[b,t], c] = acc+bias+x
// Block = 128 rows x 64 cols (N=256 exact, no col guard needed).
// ---------------------------------------------------------------------------
__global__ __launch_bounds__(256) void gemm_scatter(
    const unsigned short* __restrict__ Ahi, const unsigned short* __restrict__ Alo,
    const unsigned short* __restrict__ Bhi, const unsigned short* __restrict__ Blo,
    const float* __restrict__ bias, const int* __restrict__ ch_idx,
    const float* __restrict__ x, float* __restrict__ out) {
    int w = threadIdx.x >> 5, lane = threadIdx.x & 31;
    int wr = w >> 1, wc = w & 1;
    int hf = lane >> 4, li = lane & 15;
    int r0 = blockIdx.y * 128 + wr * 32;
    int c0 = blockIdx.x * 64 + wc * 32;
    v8f acc[2][2] = {};
    gemm_mainloop_2x2(Ahi, Alo, EP, Bhi, Blo, EP, r0, c0, li, hf, EP, acc);
#pragma unroll
    for (int i = 0; i < 2; ++i)
#pragma unroll
        for (int j = 0; j < 2; ++j) {
            int col = c0 + j * 16 + li;
            float bc = bias[col];
#pragma unroll
            for (int r = 0; r < 8; ++r) {
                int row = r0 + i * 16 + hf * 8 + r;        // in [0, MR2)
                int b = row / KCH, t = row % KCH;
                int l = ch_idx[(size_t)b * KCH + t];
                size_t o = ((size_t)b * LL + l) * DD + col;
                out[o] = acc[i][j][r] + bc + x[o];
            }
        }
}

// ---------------------------------------------------------------------------
// Host launcher
// ---------------------------------------------------------------------------
extern "C" void kernel_launch(void* const* d_in, const int* in_sizes, int n_in,
                              void* d_out, int out_size, void* d_ws, size_t ws_size,
                              hipStream_t stream) {
    (void)in_sizes; (void)n_in; (void)out_size; (void)ws_size;
    const float* x     = (const float*)d_in[0];
    const float* alpha = (const float*)d_in[1];
    const float* dyt_w = (const float*)d_in[2];
    const float* dyt_b = (const float*)d_in[3];
    const float* W_in  = (const float*)d_in[4];
    const float* b_in  = (const float*)d_in[5];
    const float* W_out = (const float*)d_in[6];
    const float* b_out = (const float*)d_in[7];
    const float* Amat  = (const float*)d_in[8];
    const float* Bp    = (const float*)d_in[9];
    const float* Cp    = (const float*)d_in[10];
    const float* convw = (const float*)d_in[11];
    const float* W_qkv = (const float*)d_in[12];
    const float* b_qkv = (const float*)d_in[13];
    float* out = (float*)d_out;

    const int NP1 = 384;    // GEMM1 padded N (336 -> 384)
    const int NP2 = 1024;   // QKV padded N (1008 -> 1024)

    // ---- workspace carve (assumes ws_size >= ~1.05 GB; qkv dominates) ----
    char* base = (char*)d_ws;
    size_t off = 0;
    auto carve = [&](size_t bytes) -> char* {
        char* p = base + off;
        off += (bytes + 255) & ~(size_t)255;
        return p;
    };
    unsigned short* xn_hi  = (unsigned short*)carve((size_t)NTOK * DD * 2);
    unsigned short* xn_lo  = (unsigned short*)carve((size_t)NTOK * DD * 2);
    unsigned short* win_hi = (unsigned short*)carve((size_t)NP1 * DD * 2);
    unsigned short* win_lo = (unsigned short*)carve((size_t)NP1 * DD * 2);
    unsigned short* wq_hi  = (unsigned short*)carve((size_t)NP2 * EP * 2);
    unsigned short* wq_lo  = (unsigned short*)carve((size_t)NP2 * EP * 2);
    unsigned short* wo_hi  = (unsigned short*)carve((size_t)DD * EP * 2);
    unsigned short* wo_lo  = (unsigned short*)carve((size_t)DD * EP * 2);
    float* xp_raw          = (float*)carve((size_t)NTOK * EE * 4);
    unsigned short* xp_hi  = (unsigned short*)carve((size_t)NTOK * EP * 2);
    unsigned short* xp_lo  = (unsigned short*)carve((size_t)NTOK * EP * 2);
    float* qkv             = (float*)carve((size_t)NTOK * Q3 * 4);
    float* imp             = (float*)carve((size_t)NTOK * 4);
    int*   ch_idx          = (int*)carve((size_t)BB * KCH * 4);
    // dead-buffer reuse (stream-ordered, no overlap hazards):
    float* attn            = xp_raw;                       // dead after norm_split
    float* xc              = (float*)xn_hi;                // dead after GEMM1 (52.8 <= 67.1 MB)
    unsigned short* o_hi   = xn_lo;                        // dead after GEMM1
    unsigned short* o_lo   = xn_lo + (size_t)MR2 * EP;     // 2*27.7 <= 67.1 MB

    // 1) pack weights to hi/lo bf16; pad N to block multiple, K to 32 multiple
    pack_w<<<(NP1 * DD + 255) / 256, 256, 0, stream>>>(W_in, win_hi, win_lo, EE, NP1, DD, DD);
    pack_w<<<(NP2 * EP + 255) / 256, 256, 0, stream>>>(W_qkv, wq_hi, wq_lo, Q3, NP2, EE, EP);
    pack_w<<<(DD * EP + 255) / 256, 256, 0, stream>>>(W_out, wo_hi, wo_lo, DD, DD, EE, EP);

    // 2) DyT -> hi/lo bf16
    dyt_split<<<(NTOK * DD) / 256, 256, 0, stream>>>(x, alpha, dyt_w, dyt_b, xn_hi, xn_lo);

    // 3) GEMM1: xp_raw = xn @ W_in^T + b_in     (131072x256 @ 256x336)
    {
        dim3 g(NP1 / 64, NTOK / 128);
        gemm_bias_f32<<<g, 256, 0, stream>>>(xn_hi, xn_lo, DD, win_hi, win_lo, DD,
                                             b_in, xp_raw, EE, EE, DD);
    }

    // 4) row L2-normalize -> hi/lo bf16 (padded EP)
    norm_split<<<NTOK / 8, 256, 0, stream>>>(xp_raw, xp_hi, xp_lo);

    // 5) QKV GEMM: qkv = xp @ W_qkv^T + b_qkv   (131072x336 @ 336x1008)
    {
        dim3 g(NP2 / 64, NTOK / 128);
        gemm_bias_f32<<<g, 256, 0, stream>>>(xp_hi, xp_lo, EP, wq_hi, wq_lo, EP,
                                             b_qkv, qkv, Q3, Q3, EP);
    }

    // 6) head top-1 select + importance
    select_heads<<<NTOK / 256, 256, 0, stream>>>(qkv, attn, imp);

    // 7) per-batch sorted top-K_CH channel indices
    topk_sort<<<BB, 1024, 0, stream>>>(imp, ch_idx);

    // 8) gather + depthwise conv
    {
        dim3 g(KCH, BB);
        gather_conv<<<g, 352, 0, stream>>>(attn, ch_idx, convw, xc);
    }

    // 9) SSM scan -> outs hi/lo (padded EP)
    ssm_scan<<<(BB * EP + 255) / 256, 256, 0, stream>>>(xc, Amat, Bp, Cp, o_hi, o_lo);

    // 10) residual copy + fused output GEMM/scatter
    copy_residual<<<(NTOK * DD) / 256, 256, 0, stream>>>(x, out);
    {
        dim3 g(DD / 64, MR2 / 128);
        gemm_scatter<<<g, 256, 0, stream>>>(o_hi, o_lo, wo_hi, wo_lo, b_out, ch_idx, x, out);
    }
}
